// SixConv_14242111553630
// MI455X (gfx1250) — compile-verified, matrix operand order checked by reading, not run on hardware
//
#include <hip/hip_runtime.h>
#include <hip/hip_bf16.h>

typedef __attribute__((ext_vector_type(16))) _Float16 v16h;
typedef __attribute__((ext_vector_type(2)))  _Float16 v2h;
typedef __attribute__((ext_vector_type(8)))  float    v8f;

#define EPS 1e-5f

// v_cvt_pk_f16_f32: pack two f32 -> two f16 (round toward zero)
__device__ __forceinline__ v2h pkh(float x, float y) {
  return __builtin_bit_cast(v2h, __builtin_amdgcn_cvt_pkrtz(x, y));
}

// ---------------------------------------------------------------------------
// Utility kernels
// ---------------------------------------------------------------------------
__global__ void zero_f32(float* p, long n) {
  long i = (long)blockIdx.x * blockDim.x + threadIdx.x;
  if (i < n) p[i] = 0.0f;
}

__global__ void degree_kernel(const int* __restrict__ dst, float* __restrict__ deg, int e) {
  int i = blockIdx.x * blockDim.x + threadIdx.x;
  if (i < e) atomicAdd(&deg[dst[i]], 1.0f);
}

__global__ void invdeg_kernel(float* deg, int n) {
  int i = blockIdx.x * blockDim.x + threadIdx.x;
  if (i < n) deg[i] = 1.0f / (deg[i] + 1.0f);   // +1 for self loop; always >= 1
}

// xu[i][h] = sum_k x[i][k] * u[k][h]   (H = 4)
__global__ void xu4_kernel(const float* __restrict__ x, const float* __restrict__ u,
                           float* __restrict__ xu, int n) {
  int i = blockIdx.x * blockDim.x + threadIdx.x;
  if (i >= n) return;
  const float* xr = x + (size_t)i * 16;
  float s0 = 0.f, s1 = 0.f, s2 = 0.f, s3 = 0.f;
#pragma unroll
  for (int k = 0; k < 16; ++k) {
    float xv = xr[k];
    const float* ur = u + k * 4;
    s0 += xv * ur[0]; s1 += xv * ur[1]; s2 += xv * ur[2]; s3 += xv * ur[3];
  }
  float4* o = (float4*)(xu + (size_t)i * 4);
  *o = make_float4(s0, s1, s2, s3);
}

// out = agg * invdeg + bias, optional relu
__global__ void finalize_kernel(const float* __restrict__ agg, const float* __restrict__ invdeg,
                                const float* __restrict__ bias, float* __restrict__ out,
                                int n, int relu) {
  long idx = (long)blockIdx.x * blockDim.x + threadIdx.x;
  if (idx >= (long)n * 16) return;
  int i = (int)(idx >> 4), f = (int)(idx & 15);
  float v = agg[idx] * invdeg[i] + bias[f];
  if (relu) v = fmaxf(v, 0.0f);
  out[idx] = v;
}

__global__ void copy_relu_kernel(const float* __restrict__ in, float* __restrict__ out, long n) {
  long i = (long)blockIdx.x * blockDim.x + threadIdx.x;
  if (i < n) out[i] = fmaxf(in[i], 0.0f);
}

__global__ void add3_kernel(float* __restrict__ h, const float* __restrict__ a,
                            const float* __restrict__ b, long n) {
  long i = (long)blockIdx.x * blockDim.x + threadIdx.x;
  if (i < n) h[i] = h[i] + a[i] + b[i];
}

// ---------------------------------------------------------------------------
// BatchNorm over N rows of 16 features: stats (sum, sumsq) then apply
// ---------------------------------------------------------------------------
__global__ void bn_stats_kernel(const float* __restrict__ x, float* __restrict__ sums, int n) {
  __shared__ float sh[256], sh2[256];
  int tid = threadIdx.x;
  int f = tid & 15;
  int rows_per_blk = blockDim.x >> 4;
  float s = 0.f, sq = 0.f;
  for (int i = blockIdx.x * rows_per_blk + (tid >> 4); i < n; i += gridDim.x * rows_per_blk) {
    float v = x[(size_t)i * 16 + f];
    s += v; sq += v * v;
  }
  sh[tid] = s; sh2[tid] = sq;
  __syncthreads();
  for (int stride = 128; stride >= 16; stride >>= 1) {
    if (tid < stride) { sh[tid] += sh[tid + stride]; sh2[tid] += sh2[tid + stride]; }
    __syncthreads();
  }
  if (tid < 16) {
    atomicAdd(&sums[tid], sh[tid]);
    atomicAdd(&sums[16 + tid], sh2[tid]);
  }
}

__global__ void bn_apply_kernel(const float* __restrict__ x, const float* __restrict__ sums,
                                const float* __restrict__ g, const float* __restrict__ be,
                                float* __restrict__ out, int n, int relu) {
  long idx = (long)blockIdx.x * blockDim.x + threadIdx.x;
  if (idx >= (long)n * 16) return;
  int f = (int)(idx & 15);
  float inv_n = 1.0f / (float)n;
  float mu = sums[f] * inv_n;
  float var = sums[16 + f] * inv_n - mu * mu;
  float v = (x[idx] - mu) * rsqrtf(var + EPS) * g[f] + be[f];
  if (relu) v = fmaxf(v, 0.0f);
  out[idx] = v;
}

// ---------------------------------------------------------------------------
// Edge kernels: 4 threads per edge; edges e >= ne are the self-loops
// ---------------------------------------------------------------------------
__global__ void edge_h4_kernel(const int* __restrict__ src, const int* __restrict__ dst,
                               const float* __restrict__ xu, const float* __restrict__ c,
                               const float* __restrict__ xw, float* __restrict__ agg,
                               int ne, int n) {
  long t = (long)blockIdx.x * blockDim.x + threadIdx.x;
  long e = t >> 2;
  int g = (int)(t & 3);
  if (e >= (long)ne + n) return;
  int s, d;
  if (e < ne) { s = src[e]; d = dst[e]; } else { s = d = (int)(e - ne); }

  // attention: q = softmax_h( xu[s][h] - xu[d][h] + c[h] )
  const float4 us = *(const float4*)(xu + (size_t)s * 4);
  const float4 ud = *(const float4*)(xu + (size_t)d * 4);
  float q0 = us.x - ud.x + c[0];
  float q1 = us.y - ud.y + c[1];
  float q2 = us.z - ud.z + c[2];
  float q3 = us.w - ud.w + c[3];
  float m = fmaxf(fmaxf(q0, q1), fmaxf(q2, q3));
  q0 = __expf(q0 - m); q1 = __expf(q1 - m); q2 = __expf(q2 - m); q3 = __expf(q3 - m);
  float inv = 1.0f / (q0 + q1 + q2 + q3);
  q0 *= inv; q1 *= inv; q2 *= inv; q3 *= inv;

  const float* xr = xw + (size_t)s * 64 + g * 4;
  float4 r0 = *(const float4*)(xr);
  float4 r1 = *(const float4*)(xr + 16);
  float4 r2 = *(const float4*)(xr + 32);
  float4 r3 = *(const float4*)(xr + 48);
  float o0 = q0 * r0.x + q1 * r1.x + q2 * r2.x + q3 * r3.x;
  float o1 = q0 * r0.y + q1 * r1.y + q2 * r2.y + q3 * r3.y;
  float o2 = q0 * r0.z + q1 * r1.z + q2 * r2.z + q3 * r3.z;
  float o3 = q0 * r0.w + q1 * r1.w + q2 * r2.w + q3 * r3.w;
  float* ap = agg + (size_t)d * 16 + g * 4;
  atomicAdd(ap + 0, o0); atomicAdd(ap + 1, o1);
  atomicAdd(ap + 2, o2); atomicAdd(ap + 3, o3);
}

__global__ void edge_h1_kernel(const int* __restrict__ src, const int* __restrict__ dst,
                               const float* __restrict__ xw, float* __restrict__ agg,
                               int ne, int n) {
  long t = (long)blockIdx.x * blockDim.x + threadIdx.x;
  long e = t >> 2;
  int g = (int)(t & 3);
  if (e >= (long)ne + n) return;
  int s, d;
  if (e < ne) { s = src[e]; d = dst[e]; } else { s = d = (int)(e - ne); }
  float4 v = *(const float4*)(xw + (size_t)s * 16 + g * 4);
  float* ap = agg + (size_t)d * 16 + g * 4;
  atomicAdd(ap + 0, v.x); atomicAdd(ap + 1, v.y);
  atomicAdd(ap + 2, v.z); atomicAdd(ap + 3, v.w);
}

// ---------------------------------------------------------------------------
// WMMA GEMM:  Y[nrows, NCOLS*16] = act( X[nrows, KIN] @ W + bias )
// One wave computes one 16x16 tile using v_wmma_f32_16x16x32_f16.
// KIN templated so A-tile loads are branch-free float4 loads + v_cvt_pk.
// ---------------------------------------------------------------------------
template <int KIN>
__device__ __forceinline__ v16h load_A_tile(const float* __restrict__ X, int row,
                                            int kc, int lane) {
  const int kb = (lane & 16) ? 8 : 0;
  const float* xr = X + (size_t)row * KIN;
  const int k0 = kc * 32 + kb;            // a[0..7]  = xr[k0 .. k0+7]
  v16h a;
  {
    float4 p0 = *(const float4*)(xr + k0);
    float4 p1 = *(const float4*)(xr + k0 + 4);
    v2h c0 = pkh(p0.x, p0.y);
    v2h c1 = pkh(p0.z, p0.w);
    v2h c2 = pkh(p1.x, p1.y);
    v2h c3 = pkh(p1.z, p1.w);
    a[0] = c0[0]; a[1] = c0[1]; a[2] = c1[0]; a[3] = c1[1];
    a[4] = c2[0]; a[5] = c2[1]; a[6] = c3[0]; a[7] = c3[1];
  }
  if constexpr (KIN >= 32) {
    const int k1 = kc * 32 + 16 + kb;     // a[8..15] = xr[k1 .. k1+7]
    float4 p0 = *(const float4*)(xr + k1);
    float4 p1 = *(const float4*)(xr + k1 + 4);
    v2h c0 = pkh(p0.x, p0.y);
    v2h c1 = pkh(p0.z, p0.w);
    v2h c2 = pkh(p1.x, p1.y);
    v2h c3 = pkh(p1.z, p1.w);
    a[8] = c0[0]; a[9] = c0[1]; a[10] = c1[0]; a[11] = c1[1];
    a[12] = c2[0]; a[13] = c2[1]; a[14] = c3[0]; a[15] = c3[1];
  } else {
    // K = 16 padded to 32 with zeros (upper half of the x32 WMMA K range)
#pragma unroll
    for (int i = 8; i < 16; ++i) a[i] = (_Float16)0.0f;
  }
  return a;
}

// pack weight W[Kin, Nout] into B-operand tiles; grid = kchunks*ncols waves of 32
__global__ void pack_w_kernel(const float* __restrict__ W, _Float16* __restrict__ out,
                              int Kin, int Nout, int ncols) {
  int tile = blockIdx.x;
  int kc = tile / ncols, nc = tile % ncols;
  int lane = threadIdx.x;
  int ncol = nc * 16 + (lane & 15);
  int kb = (lane & 16) ? 8 : 0;
  _Float16* o = out + (size_t)tile * 512 + (size_t)lane * 16;
#pragma unroll
  for (int v = 0; v < 8; ++v) {
    int k = ((v < 4) ? (kb + 2 * v) : (16 + kb + 2 * (v - 4))) + kc * 32;
#pragma unroll
    for (int hh = 0; hh < 2; ++hh) {
      int kk = k + hh;
      float f = (kk < Kin) ? W[(size_t)kk * Nout + ncol] : 0.0f;
      o[2 * v + hh] = (_Float16)f;
    }
  }
}

template <int KIN, int NCOLS, bool RELU, bool BIAS>
__global__ void wmma_gemm_kernel(const float* __restrict__ X, const _Float16* __restrict__ Bp,
                                 const float* __restrict__ bias, float* __restrict__ Y,
                                 int nrows) {
  constexpr int NOUT = NCOLS * 16;
  constexpr int KCH = (KIN + 31) / 32;
  int wave = (int)(((long)blockIdx.x * blockDim.x + threadIdx.x) >> 5);
  int lane = threadIdx.x & 31;
  int row_tiles = (nrows + 15) >> 4;
  if (wave >= row_tiles * NCOLS) return;   // wave-uniform exit: EXEC all-ones for WMMA
  int rt = wave / NCOLS, nc = wave % NCOLS;

  int row = rt * 16 + (lane & 15);
  if (row >= nrows) row = nrows - 1;       // clamp loads; store tail handled below

  v8f acc = {};
#pragma unroll
  for (int kc = 0; kc < KCH; ++kc) {
    v16h a = load_A_tile<KIN>(X, row, kc, lane);
    v16h b = *(const v16h*)(Bp + ((size_t)(kc * NCOLS + nc) * 512) + (size_t)lane * 16);
    acc = __builtin_amdgcn_wmma_f32_16x16x32_f16(false, a, false, b, (short)0, acc, false, false);
  }

  int n = lane & 15;
  int mbase = (lane & 16) ? 8 : 0;
  float bv = 0.0f;
  if constexpr (BIAS) bv = bias[nc * 16 + n];
  float* yp = Y + (size_t)(rt * 16 + mbase) * NOUT + nc * 16 + n;

  // wave-uniform fast path: tile fully in-range (always true when nrows % 16 == 0)
  if (rt * 16 + 16 <= nrows) {
#pragma unroll
    for (int j = 0; j < 8; ++j) {
      float v = acc[j];
      if constexpr (BIAS) v += bv;
      if constexpr (RELU) v = fmaxf(v, 0.0f);
      yp[(size_t)j * NOUT] = v;
    }
  } else {
#pragma unroll
    for (int j = 0; j < 8; ++j) {
      if (rt * 16 + mbase + j < nrows) {
        float v = acc[j];
        if constexpr (BIAS) v += bv;
        if constexpr (RELU) v = fmaxf(v, 0.0f);
        yp[(size_t)j * NOUT] = v;
      }
    }
  }
}

// out[i] = sigmoid( h[i,:] . ow + ob )
__global__ void final_out_kernel(const float* __restrict__ h, const float* __restrict__ ow,
                                 const float* __restrict__ ob, float* __restrict__ out, int n) {
  int i = blockIdx.x * blockDim.x + threadIdx.x;
  if (i >= n) return;
  const float* hr = h + (size_t)i * 16;
  float s = ob[0];
#pragma unroll
  for (int k = 0; k < 16; ++k) s += hr[k] * ow[k];
  out[i] = 1.0f / (1.0f + __expf(-s));
}

// ---------------------------------------------------------------------------
// Host orchestration
// ---------------------------------------------------------------------------
static inline int cdiv(long a, long b) { return (int)((a + b - 1) / b); }

extern "C" void kernel_launch(void* const* d_in, const int* in_sizes, int n_in,
                              void* d_out, int out_size, void* d_ws, size_t ws_size,
                              hipStream_t stream) {
  (void)n_in; (void)out_size; (void)ws_size;

  const float* x = (const float*)d_in[0];
  const int N = in_sizes[0] / 16;
  const int E = in_sizes[1] / 2;
  const int* src = (const int*)d_in[1];
  const int* dstp = src + E;

  // conv params: W,u,c,b for layers 1..6 at indices 2..25
  const float* W[6]; const float* U[6]; const float* C[6]; const float* B[6];
  for (int l = 0; l < 6; ++l) {
    W[l] = (const float*)d_in[2 + 4 * l + 0];
    U[l] = (const float*)d_in[2 + 4 * l + 1];
    C[l] = (const float*)d_in[2 + 4 * l + 2];
    B[l] = (const float*)d_in[2 + 4 * l + 3];
  }
  const float* g1 = (const float*)d_in[26]; const float* be1 = (const float*)d_in[27];
  const float* g2 = (const float*)d_in[28]; const float* be2 = (const float*)d_in[29];
  const float* lw1 = (const float*)d_in[30]; const float* lb1 = (const float*)d_in[31];
  const float* lw2 = (const float*)d_in[32]; const float* lb2 = (const float*)d_in[33];
  const float* lw3 = (const float*)d_in[34]; const float* lb3 = (const float*)d_in[35];
  const float* ow  = (const float*)d_in[36]; const float* ob  = (const float*)d_in[37];
  float* out = (float*)d_out;

  // ---- workspace layout ----
  _Float16* wpk = (_Float16*)d_ws;                       // packed weight tiles (29 KiB)
  float* bns = (float*)((char*)d_ws + 65536);            // BN stats: 32 floats
  float* F   = (float*)((char*)d_ws + 131072);           // float buffers
  float* deg  = F;                          // N      (becomes invdeg)
  float* xu   = F + (size_t)N;              // 4N
  float* h    = F + (size_t)5  * N;         // 16N
  float* h2   = F + (size_t)21 * N;         // 16N
  float* h4   = F + (size_t)37 * N;         // 16N
  float* tmp  = F + (size_t)53 * N;         // 16N
  float* agg  = F + (size_t)69 * N;         // 16N
  float* xw   = F + (size_t)85 * N;         // 64N (also MLP m1)
  float* m2   = h2;                         // 64N aliasing h2..agg (dead during MLP)

  // packed tile bases (tile = 512 halves)
  _Float16* Wp[6]; _Float16* L1p; _Float16* L2p; _Float16* L3p;
  Wp[0] = wpk + 0 * 512;  Wp[1] = wpk + 4 * 512;  Wp[2] = wpk + 8 * 512;
  Wp[3] = wpk + 12 * 512; Wp[4] = wpk + 13 * 512; Wp[5] = wpk + 14 * 512;
  L1p = wpk + 15 * 512; L2p = wpk + 19 * 512; L3p = wpk + 27 * 512;

  const long n16 = (long)N * 16;
  const int BLK = 256;
  const int gE16 = cdiv(n16, BLK);
  const long edgeT = ((long)E + N) * 4;
  const int gEdge = cdiv(edgeT, BLK);
  const int rowT = (N + 15) / 16;
  const int gW4 = cdiv((long)rowT * 4 * 32, BLK);   // Nout=64 gemms
  const int gW1 = cdiv((long)rowT * 1 * 32, BLK);   // Nout=16 gemms

  // ---- pack weights (tiny) ----
  for (int l = 0; l < 3; ++l)
    pack_w_kernel<<<4, 32, 0, stream>>>(W[l], Wp[l], 16, 64, 4);
  for (int l = 3; l < 6; ++l)
    pack_w_kernel<<<1, 32, 0, stream>>>(W[l], Wp[l], 16, 16, 1);
  pack_w_kernel<<<4, 32, 0, stream>>>(lw1, L1p, 16, 64, 4);
  pack_w_kernel<<<8, 32, 0, stream>>>(lw2, L2p, 64, 64, 4);
  pack_w_kernel<<<2, 32, 0, stream>>>(lw3, L3p, 64, 16, 1);

  // ---- degrees ----
  zero_f32<<<cdiv(N, BLK), BLK, 0, stream>>>(deg, N);
  degree_kernel<<<cdiv(E, BLK), BLK, 0, stream>>>(dstp, deg, E);
  invdeg_kernel<<<cdiv(N, BLK), BLK, 0, stream>>>(deg, N);

  // ---- conv layer 1 (H=4): h = relu(feast(x)) ----
  wmma_gemm_kernel<16, 4, false, false><<<gW4, BLK, 0, stream>>>(x, Wp[0], nullptr, xw, N);
  xu4_kernel<<<cdiv(N, BLK), BLK, 0, stream>>>(x, U[0], xu, N);
  zero_f32<<<gE16, BLK, 0, stream>>>(agg, n16);
  edge_h4_kernel<<<gEdge, BLK, 0, stream>>>(src, dstp, xu, C[0], xw, agg, E, N);
  finalize_kernel<<<gE16, BLK, 0, stream>>>(agg, deg, B[0], h, N, 1);

  // ---- conv layer 2 (H=4): h2 = feast(h); h = relu(h2) ----
  wmma_gemm_kernel<16, 4, false, false><<<gW4, BLK, 0, stream>>>(h, Wp[1], nullptr, xw, N);
  xu4_kernel<<<cdiv(N, BLK), BLK, 0, stream>>>(h, U[1], xu, N);
  zero_f32<<<gE16, BLK, 0, stream>>>(agg, n16);
  edge_h4_kernel<<<gEdge, BLK, 0, stream>>>(src, dstp, xu, C[1], xw, agg, E, N);
  finalize_kernel<<<gE16, BLK, 0, stream>>>(agg, deg, B[1], h2, N, 0);
  copy_relu_kernel<<<gE16, BLK, 0, stream>>>(h2, h, n16);

  // ---- conv layer 3 (H=4) + BN1 + relu ----
  wmma_gemm_kernel<16, 4, false, false><<<gW4, BLK, 0, stream>>>(h, Wp[2], nullptr, xw, N);
  xu4_kernel<<<cdiv(N, BLK), BLK, 0, stream>>>(h, U[2], xu, N);
  zero_f32<<<gE16, BLK, 0, stream>>>(agg, n16);
  edge_h4_kernel<<<gEdge, BLK, 0, stream>>>(src, dstp, xu, C[2], xw, agg, E, N);
  finalize_kernel<<<gE16, BLK, 0, stream>>>(agg, deg, B[2], tmp, N, 0);
  zero_f32<<<1, 32, 0, stream>>>(bns, 32);
  bn_stats_kernel<<<256, 256, 0, stream>>>(tmp, bns, N);
  bn_apply_kernel<<<gE16, BLK, 0, stream>>>(tmp, bns, g1, be1, h, N, 1);

  // ---- conv layer 4 (H=1): h4 = relu(feast(h)) ----
  wmma_gemm_kernel<16, 1, false, false><<<gW1, BLK, 0, stream>>>(h, Wp[3], nullptr, xw, N);
  zero_f32<<<gE16, BLK, 0, stream>>>(agg, n16);
  edge_h1_kernel<<<gEdge, BLK, 0, stream>>>(src, dstp, xw, agg, E, N);
  finalize_kernel<<<gE16, BLK, 0, stream>>>(agg, deg, B[3], h4, N, 1);

  // ---- conv layer 5 (H=1): h = relu(feast(h4)) ----
  wmma_gemm_kernel<16, 1, false, false><<<gW1, BLK, 0, stream>>>(h4, Wp[4], nullptr, xw, N);
  zero_f32<<<gE16, BLK, 0, stream>>>(agg, n16);
  edge_h1_kernel<<<gEdge, BLK, 0, stream>>>(src, dstp, xw, agg, E, N);
  finalize_kernel<<<gE16, BLK, 0, stream>>>(agg, deg, B[4], h, N, 1);

  // ---- conv layer 6 (H=1) + BN2 + relu ----
  wmma_gemm_kernel<16, 1, false, false><<<gW1, BLK, 0, stream>>>(h, Wp[5], nullptr, xw, N);
  zero_f32<<<gE16, BLK, 0, stream>>>(agg, n16);
  edge_h1_kernel<<<gEdge, BLK, 0, stream>>>(src, dstp, xw, agg, E, N);
  finalize_kernel<<<gE16, BLK, 0, stream>>>(agg, deg, B[5], tmp, N, 0);
  zero_f32<<<1, 32, 0, stream>>>(bns, 32);
  bn_stats_kernel<<<256, 256, 0, stream>>>(tmp, bns, N);
  bn_apply_kernel<<<gE16, BLK, 0, stream>>>(tmp, bns, g2, be2, h, N, 1);

  // ---- residual: h += h2 + h4 ----
  add3_kernel<<<gE16, BLK, 0, stream>>>(h, h2, h4, n16);

  // ---- MLP head (all WMMA) ----
  wmma_gemm_kernel<16, 4, true, true><<<gW4, BLK, 0, stream>>>(h, L1p, lb1, xw, N);   // 16->64
  wmma_gemm_kernel<64, 4, true, true><<<gW4, BLK, 0, stream>>>(xw, L2p, lb2, m2, N);  // 64->64
  wmma_gemm_kernel<64, 1, true, true><<<gW1, BLK, 0, stream>>>(m2, L3p, lb3, h, N);   // 64->16
  final_out_kernel<<<cdiv(N, BLK), BLK, 0, stream>>>(h, ow, ob, out, N);              // 16->1 sigmoid
}
